// MultiHeadAttention_1726576856575
// MI455X (gfx1250) — compile-verified, matrix operand order checked by reading, not run on hardware
//
#include <hip/hip_runtime.h>
#include <hip/hip_bf16.h>
#include <stdint.h>

#define N_HEADS 16
#define DMODEL  1024
#define DK      64
#define DV      64
#define BATCH   2
#define SEQ     2048
#define BSTOK   (BATCH*SEQ)   /* 4096 token rows */

typedef __bf16 bf16;
typedef __attribute__((ext_vector_type(16))) __bf16 v16bf;
typedef __attribute__((ext_vector_type(8)))  __bf16 v8bf;
typedef __attribute__((ext_vector_type(4)))  __bf16 v4bf;
typedef __attribute__((ext_vector_type(8)))  float  v8f;

// Per-lane 16-element WMMA operand from two contiguous 16-byte chunks,
// matching the CDNA5 16-bit A/B lane layout:
//   lanes 0-15 : K = base+0..7   and base+16..23
//   lanes 16-31: K = base+8..15  and base+24..31   (caller passes khalf=8)
__device__ __forceinline__ v16bf pack16(const bf16* p0, const bf16* p1) {
    v8bf a0 = *(const v8bf*)p0;
    v8bf a1 = *(const v8bf*)p1;
    v16bf r;
#pragma unroll
    for (int i = 0; i < 8; ++i) { r[i] = a0[i]; r[i + 8] = a1[i]; }
    return r;
}

// ---------------- elementwise f32 -> bf16 cast ----------------
__global__ __launch_bounds__(256) void cast_f32_bf16(const float* __restrict__ x,
                                                     bf16* __restrict__ y, int n) {
    int i = blockIdx.x * 256 + threadIdx.x;
    if (i < n) y[i] = (bf16)x[i];
}

// ---------------- layernorm (pre-LN on q) -> bf16 ----------------
__global__ __launch_bounds__(256) void layernorm_to_bf16(const float* __restrict__ x,
                                                         const float* __restrict__ g,
                                                         const float* __restrict__ b,
                                                         bf16* __restrict__ y) {
    __shared__ float red[256];
    const int row = blockIdx.x;
    const float* xr = x + (size_t)row * DMODEL;
    float s = 0.f;
    for (int c = threadIdx.x; c < DMODEL; c += 256) s += xr[c];
    red[threadIdx.x] = s; __syncthreads();
    for (int off = 128; off; off >>= 1) {
        if (threadIdx.x < off) red[threadIdx.x] += red[threadIdx.x + off];
        __syncthreads();
    }
    const float mu = red[0] * (1.0f / DMODEL);
    __syncthreads();
    float v = 0.f;
    for (int c = threadIdx.x; c < DMODEL; c += 256) { float d = xr[c] - mu; v += d * d; }
    red[threadIdx.x] = v; __syncthreads();
    for (int off = 128; off; off >>= 1) {
        if (threadIdx.x < off) red[threadIdx.x] += red[threadIdx.x + off];
        __syncthreads();
    }
    const float rstd = rsqrtf(red[0] * (1.0f / DMODEL) + 1e-6f);
    bf16* yr = y + (size_t)row * DMODEL;
    for (int c = threadIdx.x; c < DMODEL; c += 256)
        yr[c] = (bf16)((xr[c] - mu) * rstd * g[c] + b[c]);
}

// ---------------- GEMM: Y[M,N] = X[M,K] * W[N,K]^T  (torch Linear layout) ---
// One 32x32 macro-tile per wave: 2 A-loads + 2 B-loads feed 4 WMMAs per K-step.
template <bool FC_EPILOGUE>
__global__ __launch_bounds__(256) void gemm_bf16_wmma(const bf16* __restrict__ X,
                                                      const bf16* __restrict__ W,
                                                      int M, int N, int K,
                                                      bf16* __restrict__ Ybf,
                                                      float* __restrict__ Yf,
                                                      const float* __restrict__ bias,
                                                      const float* __restrict__ resid) {
    const int tilesN = N >> 5;                 // 32-wide macro tiles
    const int total  = (M >> 5) * tilesN;
    const int wid = blockIdx.x * (blockDim.x >> 5) + (threadIdx.x >> 5);
    if (wid >= total) return;                  // wave-uniform: EXEC stays all-1s
    const int tm = wid / tilesN;
    const int tn = wid - tm * tilesN;
    const int lane  = threadIdx.x & 31;
    const int l15   = lane & 15;
    const int khalf = (lane >> 4) << 3;

    const bf16* xr0 = X + (size_t)(tm * 32 + l15) * K;
    const bf16* xr1 = xr0 + (size_t)16 * K;
    const bf16* wr0 = W + (size_t)(tn * 32 + l15) * K;
    const bf16* wr1 = wr0 + (size_t)16 * K;
    v8f d00 = {}, d01 = {}, d10 = {}, d11 = {};
    for (int kk = 0; kk < K; kk += 32) {
        __builtin_prefetch(xr0 + kk + 256, 0, 1);
        __builtin_prefetch(wr0 + kk + 256, 0, 1);
        v16bf a0 = pack16(xr0 + kk + khalf, xr0 + kk + 16 + khalf);
        v16bf a1 = pack16(xr1 + kk + khalf, xr1 + kk + 16 + khalf);
        v16bf b0 = pack16(wr0 + kk + khalf, wr0 + kk + 16 + khalf);
        v16bf b1 = pack16(wr1 + kk + khalf, wr1 + kk + 16 + khalf);
        d00 = __builtin_amdgcn_wmma_f32_16x16x32_bf16(false, a0, false, b0, (short)0, d00, false, false);
        d01 = __builtin_amdgcn_wmma_f32_16x16x32_bf16(false, a0, false, b1, (short)0, d01, false, false);
        d10 = __builtin_amdgcn_wmma_f32_16x16x32_bf16(false, a1, false, b0, (short)0, d10, false, false);
        d11 = __builtin_amdgcn_wmma_f32_16x16x32_bf16(false, a1, false, b1, (short)0, d11, false, false);
    }
    // C/D layout: VGPR r -> M = r (lanes 0-15) or r+8 (lanes 16-31); N = lane&15
    const int mb = (lane >> 4) << 3;
    const int m0 = tm * 32 + mb, m1 = m0 + 16;
    const int n0 = tn * 32 + l15, n1 = n0 + 16;
#pragma unroll
    for (int r = 0; r < 8; ++r) {
        const size_t i00 = (size_t)(m0 + r) * N + n0;
        const size_t i01 = (size_t)(m0 + r) * N + n1;
        const size_t i10 = (size_t)(m1 + r) * N + n0;
        const size_t i11 = (size_t)(m1 + r) * N + n1;
        if (FC_EPILOGUE) {
            Yf[i00] = d00[r] + bias[n0] + resid[i00];
            Yf[i01] = d01[r] + bias[n1] + resid[i01];
            Yf[i10] = d10[r] + bias[n0] + resid[i10];
            Yf[i11] = d11[r] + bias[n1] + resid[i11];
        } else {
            Ybf[i00] = (bf16)d00[r];
            Ybf[i01] = (bf16)d01[r];
            Ybf[i10] = (bf16)d10[r];
            Ybf[i11] = (bf16)d11[r];
        }
    }
}

// ---------------- V head transpose: [BS, H*64] -> [B,H,64,S] ----------------
__global__ __launch_bounds__(256) void transpose_v(const bf16* __restrict__ Vh,
                                                   bf16* __restrict__ Vt) {
    int i = blockIdx.x * 256 + threadIdx.x;
    if (i >= BSTOK * DMODEL) return;
    const int sglob = i / DMODEL;
    const int n = i - sglob * DMODEL;
    const int b = sglob / SEQ;
    const int sl = sglob - b * SEQ;
    const int h = n >> 6;
    const int d = n & 63;
    Vt[((size_t)(b * N_HEADS + h) * DV + d) * SEQ + sl] = Vh[i];
}

// ---------------- attention: scores -> softmax -> attn out -> P@V ----------
__global__ __launch_bounds__(256) void attention_kernel(const bf16* __restrict__ Qh,
                                                        const bf16* __restrict__ Kh,
                                                        const bf16* __restrict__ Vt,
                                                        const uint8_t* __restrict__ mask,
                                                        float* __restrict__ attn,
                                                        bf16* __restrict__ HO) {
    extern __shared__ char smem[];
    float* sS   = (float*)smem;                            // [16][SEQ] f32 scores
    bf16*  sP   = (bf16*)(smem + 16 * SEQ * 4);            // [16][SEQ] bf16 probs
    float* sRed = (float*)(smem + 16 * SEQ * 4 + 16 * SEQ * 2); // 256 partials
    float* sMax = sRed + 256;                              // 16 row maxima
    float* sInv = sMax + 16;                               // 16 reciprocal sums

    const int qt = blockIdx.x & 127;           // SEQ/16 = 128 q-tiles
    const int h  = (blockIdx.x >> 7) & 15;
    const int b  = blockIdx.x >> 11;
    const int wave  = threadIdx.x >> 5;
    const int lane  = threadIdx.x & 31;
    const int l15   = lane & 15;
    const int khalf = (lane >> 4) << 3;

    // -- scores: 8 waves cover 128 16x16 column tiles, K=64 per head --
    const bf16* qrow = Qh + (size_t)(b * SEQ + qt * 16 + l15) * DMODEL + h * DK;
    for (int ct = wave; ct < SEQ / 16; ct += 8) {
        const bf16* krow = Kh + (size_t)(b * SEQ + ct * 16 + l15) * DMODEL + h * DK;
        v8f acc = {};
#pragma unroll
        for (int kk = 0; kk < DK; kk += 32) {
            v16bf a  = pack16(qrow + kk + khalf, qrow + kk + 16 + khalf);
            v16bf bb = pack16(krow + kk + khalf, krow + kk + 16 + khalf);
            acc = __builtin_amdgcn_wmma_f32_16x16x32_bf16(false, a, false, bb,
                                                          (short)0, acc, false, false);
        }
        const int mb = (lane >> 4) << 3;
#pragma unroll
        for (int r = 0; r < 8; ++r)
            sS[(mb + r) * SEQ + ct * 16 + l15] = acc[r];
    }
    __syncthreads();

    // -- softmax pass 1: mask + scale + row max (16 threads/row, float4 regs) --
    const int row = threadIdx.x >> 4;          // 0..15
    const int lr  = threadIdx.x & 15;
    const int sq  = b * SEQ + qt * 16 + row;
    float4* srow4 = (float4*)(sS + row * SEQ);
    const uint32_t* m4row = (const uint32_t*)(mask + (size_t)sq * SEQ);

    float pmax = -3.0e38f;
    for (int j = 0; j < SEQ / 64; ++j) {       // 512 quads/row over 16 threads
        const int qd = j * 16 + lr;
        float4 sv = srow4[qd];                 // ds_load_b128
        const uint32_t m4 = m4row[qd];         // 4 mask bytes, coalesced
        sv.x = (m4 & 0x000000ffu) ? -1e9f : sv.x * 0.125f;   // 1/sqrt(64)
        sv.y = (m4 & 0x0000ff00u) ? -1e9f : sv.y * 0.125f;
        sv.z = (m4 & 0x00ff0000u) ? -1e9f : sv.z * 0.125f;
        sv.w = (m4 & 0xff000000u) ? -1e9f : sv.w * 0.125f;
        srow4[qd] = sv;
        pmax = fmaxf(fmaxf(pmax, fmaxf(sv.x, sv.y)), fmaxf(sv.z, sv.w));
    }
    sRed[threadIdx.x] = pmax; __syncthreads();
    if (threadIdx.x < 16) {
        float m = sRed[threadIdx.x * 16];
#pragma unroll
        for (int i = 1; i < 16; ++i) m = fmaxf(m, sRed[threadIdx.x * 16 + i]);
        sMax[threadIdx.x] = m;
    }
    __syncthreads();

    // -- pass 2: exp + row sum --
    const float rmax = sMax[row];
    float psum = 0.f;
    for (int j = 0; j < SEQ / 64; ++j) {
        const int qd = j * 16 + lr;
        float4 sv = srow4[qd];
        sv.x = __expf(sv.x - rmax);
        sv.y = __expf(sv.y - rmax);
        sv.z = __expf(sv.z - rmax);
        sv.w = __expf(sv.w - rmax);
        srow4[qd] = sv;
        psum += (sv.x + sv.y) + (sv.z + sv.w);
    }
    sRed[threadIdx.x] = psum; __syncthreads();
    if (threadIdx.x < 16) {
        float s = 0.f;
#pragma unroll
        for (int i = 0; i < 16; ++i) s += sRed[threadIdx.x * 16 + i];
        sInv[threadIdx.x] = 1.0f / s;
    }
    __syncthreads();

    // -- pass 3: normalize + stores, FLAT mapping so 32 lanes write 512B runs --
    {
        float4* sS4 = (float4*)sS;
        v4bf*   sP4 = (v4bf*)sP;
        float4* a4  = (float4*)(attn + ((size_t)(b * N_HEADS + h) * SEQ + qt * 16) * SEQ);
        for (int i = 0; i < 32; ++i) {         // 8192 quads over 256 threads
            const int qd = i * 256 + threadIdx.x;
            const float rinv = sInv[qd >> 9];  // 512 quads per row
            float4 pv = sS4[qd];
            pv.x *= rinv; pv.y *= rinv; pv.z *= rinv; pv.w *= rinv;
            a4[qd] = pv;                       // dominant HBM store, fully coalesced
            v4bf pb; pb[0] = (bf16)pv.x; pb[1] = (bf16)pv.y;
            pb[2] = (bf16)pv.z; pb[3] = (bf16)pv.w;
            sP4[qd] = pb;
        }
    }
    __syncthreads();

    // -- P @ V : all 8 waves; (n-tile, K-half) split, LDS partial reduction --
    const int nt   = wave & 3;                 // which 16 output dims
    const int ksel = wave >> 2;                // which half of K=SEQ
    const bf16* pr = sP + (size_t)l15 * SEQ + ksel * (SEQ / 2);
    const bf16* vr = Vt + ((size_t)(b * N_HEADS + h) * DV + nt * 16 + l15) * SEQ
                        + ksel * (SEQ / 2);
    v8f acc = {};
    for (int kk = 0; kk < SEQ / 2; kk += 32) {
        v16bf a  = pack16(pr + kk + khalf, pr + kk + 16 + khalf);   // ds_load_b128
        v16bf bb = pack16(vr + kk + khalf, vr + kk + 16 + khalf);
        acc = __builtin_amdgcn_wmma_f32_16x16x32_bf16(false, a, false, bb,
                                                      (short)0, acc, false, false);
    }
    const int mb = (lane >> 4) << 3;
    float* part = sS;                          // sS is dead now; reuse 4KB of it
    if (ksel) {
#pragma unroll
        for (int r = 0; r < 8; ++r) part[nt * 256 + (mb + r) * 16 + l15] = acc[r];
    }
    __syncthreads();
    if (!ksel) {
#pragma unroll
        for (int r = 0; r < 8; ++r) {
            const float o = acc[r] + part[nt * 256 + (mb + r) * 16 + l15];
            const int sql = qt * 16 + mb + r;
            HO[(size_t)(b * SEQ + sql) * DMODEL + h * DV + nt * 16 + l15] = (bf16)o;
        }
    }
}

// ---------------------------------------------------------------------------
extern "C" void kernel_launch(void* const* d_in, const int* in_sizes, int n_in,
                              void* d_out, int out_size, void* d_ws, size_t ws_size,
                              hipStream_t stream) {
    (void)in_sizes; (void)n_in; (void)out_size; (void)ws_size;
    const float*   q    = (const float*)d_in[0];
    const float*   k    = (const float*)d_in[1];
    const float*   v    = (const float*)d_in[2];
    const uint8_t* mask = (const uint8_t*)d_in[3];
    const float*   w_q  = (const float*)d_in[4];
    const float*   w_k  = (const float*)d_in[5];
    const float*   w_v  = (const float*)d_in[6];
    const float*   w_fc = (const float*)d_in[7];
    const float*   b_fc = (const float*)d_in[8];
    const float*   ln_g = (const float*)d_in[9];
    const float*   ln_b = (const float*)d_in[10];

    char* ws = (char*)d_ws;
    size_t o = 0;
    const size_t TOKB = (size_t)BSTOK * DMODEL * sizeof(bf16);   // 8 MB
    const size_t WB   = (size_t)DMODEL * DMODEL * sizeof(bf16);  // 2 MB
    bf16* qn   = (bf16*)(ws + o); o += TOKB;
    bf16* kb   = (bf16*)(ws + o); o += TOKB;
    bf16* vb   = (bf16*)(ws + o); o += TOKB;
    bf16* wqb  = (bf16*)(ws + o); o += WB;
    bf16* wkb  = (bf16*)(ws + o); o += WB;
    bf16* wvb  = (bf16*)(ws + o); o += WB;
    bf16* wfcb = (bf16*)(ws + o); o += WB;
    bf16* Qh   = (bf16*)(ws + o); o += TOKB;
    bf16* Kh   = (bf16*)(ws + o); o += TOKB;
    bf16* Vh   = (bf16*)(ws + o); o += TOKB;
    bf16* Vt   = (bf16*)(ws + o); o += TOKB;
    bf16* HO   = (bf16*)(ws + o); o += TOKB;   // total ~72 MB of workspace

    float* out  = (float*)d_out;
    float* attn = out + (size_t)BSTOK * DMODEL;

    const int nTok = BSTOK * DMODEL;    // 4,194,304
    const int nW   = DMODEL * DMODEL;   // 1,048,576

    cast_f32_bf16<<<nTok / 256, 256, 0, stream>>>(k, kb, nTok);
    cast_f32_bf16<<<nTok / 256, 256, 0, stream>>>(v, vb, nTok);
    cast_f32_bf16<<<nW / 256, 256, 0, stream>>>(w_q, wqb, nW);
    cast_f32_bf16<<<nW / 256, 256, 0, stream>>>(w_k, wkb, nW);
    cast_f32_bf16<<<nW / 256, 256, 0, stream>>>(w_v, wvb, nW);
    cast_f32_bf16<<<nW / 256, 256, 0, stream>>>(w_fc, wfcb, nW);
    layernorm_to_bf16<<<BSTOK, 256, 0, stream>>>(q, ln_g, ln_b, qn);

    const int tiles   = (BSTOK / 32) * (DMODEL / 32);   // 4096 macro tiles
    const int gblocks = tiles / 8;                      // 8 waves per block
    gemm_bf16_wmma<false><<<gblocks, 256, 0, stream>>>(qn, wqb, BSTOK, DMODEL, DMODEL,
                                                       Qh, nullptr, nullptr, nullptr);
    gemm_bf16_wmma<false><<<gblocks, 256, 0, stream>>>(kb, wkb, BSTOK, DMODEL, DMODEL,
                                                       Kh, nullptr, nullptr, nullptr);
    gemm_bf16_wmma<false><<<gblocks, 256, 0, stream>>>(vb, wvb, BSTOK, DMODEL, DMODEL,
                                                       Vh, nullptr, nullptr, nullptr);
    transpose_v<<<nTok / 256, 256, 0, stream>>>(Vh, Vt);

    const size_t smem = (size_t)16 * SEQ * 4 + (size_t)16 * SEQ * 2 + (256 + 32) * 4;
    attention_kernel<<<BATCH * N_HEADS * (SEQ / 16), 256, smem, stream>>>(
        Qh, Kh, Vt, mask, attn, HO);

    gemm_bf16_wmma<true><<<gblocks, 256, 0, stream>>>(HO, wfcb, BSTOK, DMODEL, DMODEL,
                                                      nullptr, out, b_fc, q);
}
